// ORNet_30846455120029
// MI455X (gfx1250) — matrix operand order repaired and checked by reading, not compiled, not run on hardware
//
#include <hip/hip_runtime.h>
#include <hip/hip_bf16.h>
#include <math.h>

typedef __attribute__((ext_vector_type(16))) _Float16 v16h;
typedef __attribute__((ext_vector_type(8)))  float    v8f;

#define NB   16      // batches
#define NP   2048    // points
#define KNN  12      // neighbors
#define EC   64      // channels
#define EPSN 1e-5f

// ---------------------------------------------------------------------------
// WMMA helpers (CDNA5 f16 -> f32, 16x16x32)
// ---------------------------------------------------------------------------
__device__ __forceinline__ v8f wmma_f16(v16h a, v16h b, v8f c) {
  return __builtin_amdgcn_wmma_f32_16x16x32_f16(
      /*neg_a=*/false, a, /*neg_b=*/false, b,
      /*c_mod=*/(short)0, c, /*reuse_a=*/false, /*reuse_b=*/false);
}

// A fragment: 16x32 tile of row-major [M][K] LDS array.
// Layout: lanes 0-15 row M=lane, elems 0-15 -> K 0-15;
// lanes 16-31 same rows, K 16-31.
__device__ __forceinline__ v16h load_fragA(const _Float16* base, int ld,
                                           int m0, int k0, int lane) {
  v16h f;
  const _Float16* r =
      base + (size_t)(m0 + (lane & 15)) * ld + k0 + ((lane >= 16) ? 16 : 0);
#pragma unroll
  for (int e = 0; e < 16; ++e) f[e] = r[e];
  return f;
}

// B fragment: 32x16 tile of row-major [K][N] LDS array.
// Layout: element e holds row K=e (lanes 0-15) / K=16+e (lanes 16-31), col = lane&15.
__device__ __forceinline__ v16h load_fragB(const _Float16* base, int ld,
                                           int k0, int n0, int lane) {
  v16h f;
  const _Float16* r =
      base + (size_t)(k0 + ((lane >= 16) ? 16 : 0)) * ld + n0 + (lane & 15);
#pragma unroll
  for (int e = 0; e < 16; ++e) f[e] = r[(size_t)e * ld];
  return f;
}

// B fragment from transposed storage: base[n*ld + k] = B[k][n] (contiguous K reads).
__device__ __forceinline__ v16h load_fragB_T(const _Float16* base, int ld,
                                             int k0, int n0, int lane) {
  v16h f;
  const _Float16* r =
      base + (size_t)(n0 + (lane & 15)) * ld + k0 + ((lane >= 16) ? 16 : 0);
#pragma unroll
  for (int e = 0; e < 16; ++e) f[e] = r[e];
  return f;
}

// ---------------------------------------------------------------------------
// KNN: one workgroup per (batch, 256-point slab); batch cloud staged in LDS.
// ---------------------------------------------------------------------------
__global__ void __launch_bounds__(256) knn_kernel(const float* __restrict__ src,
                                                  int* __restrict__ nn) {
  __shared__ float px[NP], py[NP], pz[NP];
  const int b = blockIdx.y;
  const float* xb = src + (size_t)b * 3 * NP;
  for (int i = threadIdx.x; i < NP; i += 256) {
    px[i] = xb[i]; py[i] = xb[NP + i]; pz[i] = xb[2 * NP + i];
  }
  __syncthreads();
  const int n = blockIdx.x * 256 + threadIdx.x;
  const float qx = px[n], qy = py[n], qz = pz[n];
  float best[KNN]; int besti[KNN];
#pragma unroll
  for (int j = 0; j < KNN; ++j) { best[j] = -3.4e38f; besti[j] = 0; }
  for (int m = 0; m < NP; ++m) {
    const float dx = qx - px[m], dy = qy - py[m], dz = qz - pz[m];
    const float nd = -(dx * dx + dy * dy + dz * dz);
    if (nd > best[KNN - 1]) {
      int j = KNN - 1;
      while (j > 0 && nd > best[j - 1]) {
        best[j] = best[j - 1]; besti[j] = besti[j - 1]; --j;
      }
      best[j] = nd; besti[j] = m;
    }
  }
  int* ob = nn + ((size_t)b * NP + n) * KNN;
#pragma unroll
  for (int j = 0; j < KNN; ++j) ob[j] = besti[j];
}

__global__ void zero_kernel(float* __restrict__ p, int n) {
  const int i = blockIdx.x * 256 + threadIdx.x;
  if (i < n) p[i] = 0.f;
}

// ---------------------------------------------------------------------------
// Stats: M1[c] = sum_s D[c,s], M2[c,c'] = sum_s D[c,s]D[c',s]  (WMMA rank-S
// update of diff second-moment matrix -> exact BN mean/var via linearity).
// ---------------------------------------------------------------------------
template <int CPAD>
__global__ void __launch_bounds__(128) stats_kernel(
    const float* __restrict__ x, const int* __restrict__ nn, int C,
    float* __restrict__ M1, float* __restrict__ M2) {
  constexpr int TS = 384;         // sample columns per block (divides B*N*K)
  constexpr int LDD = TS + 8;
  __shared__ _Float16 Dt[CPAD * LDD];
  const int tid = threadIdx.x, lane = tid & 31, wave = tid >> 5;
  const long s0 = (long)blockIdx.x * TS;

  for (int cl = tid; cl < TS; cl += 128) {
    const long s = s0 + cl;
    const int b = (int)(s / (NP * KNN));
    const int r = (int)(s - (long)b * NP * KNN);
    const int n = r / KNN, k = r - n * KNN;
    const int j = nn[((size_t)b * NP + n) * KNN + k];
    const float* xb = x + (size_t)b * C * NP;
    for (int c = 0; c < C; ++c)
      Dt[c * LDD + cl] = (_Float16)(xb[(size_t)c * NP + j] - xb[(size_t)c * NP + n]);
    for (int c = C; c < CPAD; ++c) Dt[c * LDD + cl] = (_Float16)0.f;
  }
  __syncthreads();

  if (tid < CPAD) {   // M1 row sums
    float s = 0.f;
    for (int t = 0; t < TS; ++t) s += (float)Dt[tid * LDD + t];
    atomicAdd(&M1[tid], s);
  }

  constexpr int NT = CPAD / 16;
  for (int tile = wave; tile < NT * NT; tile += 4) {
    const int ti = tile / NT, tj = tile - (tile / NT) * NT;
    v8f acc = {};
#pragma unroll
    for (int kk = 0; kk < TS / 32; ++kk) {
      v16h a  = load_fragA (Dt, LDD, ti * 16, kk * 32, lane);
      v16h bf = load_fragB_T(Dt, LDD, kk * 32, tj * 16, lane);
      acc = wmma_f16(a, bf, acc);
    }
    const int mo = ti * 16 + ((lane >= 16) ? 8 : 0);
    const int nc = tj * 16 + (lane & 15);
#pragma unroll
    for (int v = 0; v < 8; ++v) atomicAdd(&M2[(mo + v) * EC + nc], acc[v]);
  }
}

// scale/bias from moments: mean_o = w_o.M1/S ; E[y^2]_o = w_o^T M2 w_o / S
__global__ void finalize_stats_kernel(
    const float* __restrict__ M1, const float* __restrict__ M2,
    const float* __restrict__ w2a, const float* __restrict__ g,
    const float* __restrict__ bt, int C, float invS,
    float* __restrict__ scl, float* __restrict__ bia) {
  const int o = threadIdx.x;  // 64 threads
  const float* wo = w2a + o * C;
  float mean = 0.f;
  for (int c = 0; c < C; ++c) mean += wo[c] * M1[c];
  mean *= invS;
  float e2 = 0.f;
  for (int c = 0; c < C; ++c) {
    const float wc = wo[c];
    for (int c2 = 0; c2 < C; ++c2) e2 += wc * wo[c2] * M2[c * EC + c2];
  }
  e2 *= invS;
  const float var = e2 - mean * mean;
  const float sc = g[o] * rsqrtf(var + EPSN);
  scl[o] = sc;
  bia[o] = bt[o] - mean * sc;
}

// ---------------------------------------------------------------------------
// Fused propagate: gather-diff -> W2a GEMM -> BN+ReLU -> W2b GEMM -> max_k
// -> W1 GEMM (+biases). One WG per (batch, 16-point tile) => 192 sample cols.
// Dynamic LDS; Zt(f32) overlaps Dt(f16) lifetime (union region).
// ---------------------------------------------------------------------------
template <int CPAD>
struct PCfg {
  static constexpr int SC = 192;           // 16 points * 12 neighbors
  static constexpr int LDD = SC + 8;
  static constexpr int LDA = CPAD + 8;
  static constexpr int LDW = EC + 8;
  static constexpr size_t oWa = 0;
  static constexpr size_t oWb = oWa + (size_t)EC * LDA * 2;
  static constexpr size_t oW1 = oWb + (size_t)EC * LDW * 2;
  static constexpr size_t oYt = oW1 + (size_t)EC * LDW * 2;
  static constexpr size_t oZm = oYt + (size_t)EC * LDD * 2;
  static constexpr size_t oF  = oZm + (size_t)EC * 24 * 2;
  static constexpr size_t oU  = (oF + (size_t)4 * EC * 4 + 15) & ~(size_t)15;
  static constexpr size_t szD = (size_t)CPAD * LDD * 2;
  static constexpr size_t szZ = (size_t)EC * SC * 4;
  static constexpr size_t total = oU + (szD > szZ ? szD : szZ);
};

template <int CPAD>
__global__ void __launch_bounds__(256) propagate_kernel(
    const float* __restrict__ x, const int* __restrict__ nn,
    const float* __restrict__ w2a, const float* __restrict__ scl,
    const float* __restrict__ bia, const float* __restrict__ w2b,
    const float* __restrict__ bb, const float* __restrict__ w1,
    const float* __restrict__ b1, float* __restrict__ out, long obstride,
    int C) {
  using CF = PCfg<CPAD>;
  extern __shared__ char smem[];
  _Float16* Wa  = (_Float16*)(smem + CF::oWa);
  _Float16* Wb  = (_Float16*)(smem + CF::oWb);
  _Float16* W1s = (_Float16*)(smem + CF::oW1);
  _Float16* Yt  = (_Float16*)(smem + CF::oYt);
  _Float16* Zm  = (_Float16*)(smem + CF::oZm);
  float* sc_s = (float*)(smem + CF::oF);
  float* bi_s = sc_s + EC;
  float* bb_s = sc_s + 2 * EC;
  float* b1_s = sc_s + 3 * EC;
  _Float16* Dt = (_Float16*)(smem + CF::oU);   // alive: build -> GEMM1
  float*    Zt = (float*)(smem + CF::oU);      // alive: GEMM2 -> max

  const int tid = threadIdx.x, lane = tid & 31, wave = tid >> 5;
  const int b = blockIdx.y, n0 = blockIdx.x * 16;

  for (int i = tid; i < EC * CPAD; i += 256) {
    const int o = i / CPAD, c = i - o * CPAD;
    Wa[o * CF::LDA + c] = (c < C) ? (_Float16)w2a[o * C + c] : (_Float16)0.f;
  }
  for (int i = tid; i < EC * EC; i += 256) {
    const int o = i >> 6, c = i & 63;
    Wb[o * CF::LDW + c]  = (_Float16)w2b[i];
    W1s[o * CF::LDW + c] = (_Float16)w1[i];
  }
  if (tid < EC) {
    sc_s[tid] = scl[tid]; bi_s[tid] = bia[tid];
    bb_s[tid] = bb[tid];  b1_s[tid] = b1[tid];
  }

  const float* xb = x + (size_t)b * C * NP;
  for (int col = tid; col < CF::SC; col += 256) {
    const int nl = col / KNN, k = col - nl * KNN;
    const int n = n0 + nl;
    const int j = nn[((size_t)b * NP + n) * KNN + k];
    for (int c = 0; c < C; ++c)
      Dt[c * CF::LDD + col] =
          (_Float16)(xb[(size_t)c * NP + j] - xb[(size_t)c * NP + n]);
    for (int c = C; c < CPAD; ++c) Dt[c * CF::LDD + col] = (_Float16)0.f;
  }
  __syncthreads();

  // GEMM1: Y = ReLU(BN(W2a @ D))  -> Yt (f16). 4 o-tiles x 12 s-tiles, 8 waves.
  for (int tile = wave; tile < 48; tile += 8) {
    const int oi = tile / 12, sj = tile - (tile / 12) * 12;
    v8f acc = {};
#pragma unroll
    for (int kk = 0; kk < CPAD / 32; ++kk) {
      v16h a  = load_fragA(Wa, CF::LDA, oi * 16, kk * 32, lane);
      v16h bf = load_fragB(Dt, CF::LDD, kk * 32, sj * 16, lane);
      acc = wmma_f16(a, bf, acc);
    }
    const int mo = oi * 16 + ((lane >= 16) ? 8 : 0);
    const int nc = sj * 16 + (lane & 15);
#pragma unroll
    for (int v = 0; v < 8; ++v) {
      const int o = mo + v;
      const float y = fmaf(acc[v], sc_s[o], bi_s[o]);
      Yt[o * CF::LDD + nc] = (_Float16)fmaxf(y, 0.f);
    }
  }
  __syncthreads();

  // GEMM2: Z = W2b @ Y -> Zt (f32, overlaps Dt)
  for (int tile = wave; tile < 48; tile += 8) {
    const int oi = tile / 12, sj = tile - (tile / 12) * 12;
    v8f acc = {};
#pragma unroll
    for (int kk = 0; kk < 2; ++kk) {
      v16h a  = load_fragA(Wb, CF::LDW, oi * 16, kk * 32, lane);
      v16h bf = load_fragB(Yt, CF::LDD, kk * 32, sj * 16, lane);
      acc = wmma_f16(a, bf, acc);
    }
    const int mo = oi * 16 + ((lane >= 16) ? 8 : 0);
    const int nc = sj * 16 + (lane & 15);
#pragma unroll
    for (int v = 0; v < 8; ++v) Zt[(mo + v) * CF::SC + nc] = acc[v];
  }
  __syncthreads();

  // max over k neighbors + bb -> Zm (f16, 64 x 16)
  for (int i = tid; i < EC * 16; i += 256) {
    const int o = i >> 4, nl = i & 15;
    float m = Zt[o * CF::SC + nl * KNN];
#pragma unroll
    for (int k = 1; k < KNN; ++k) m = fmaxf(m, Zt[o * CF::SC + nl * KNN + k]);
    Zm[o * 24 + nl] = (_Float16)(m + bb_s[o]);
  }
  __syncthreads();

  // GEMM3: Xn = W1 @ Zm + b1 -> global
  if (wave < 4) {
    const int oi = wave;
    v8f acc = {};
#pragma unroll
    for (int kk = 0; kk < 2; ++kk) {
      v16h a  = load_fragA(W1s, CF::LDW, oi * 16, kk * 32, lane);
      v16h bf = load_fragB(Zm, 24, kk * 32, 0, lane);
      acc = wmma_f16(a, bf, acc);
    }
    const int mo = oi * 16 + ((lane >= 16) ? 8 : 0);
    const int nl = lane & 15;
    float* ob = out + (size_t)b * obstride;
#pragma unroll
    for (int v = 0; v < 8; ++v) {
      const int o = mo + v;
      ob[(size_t)o * NP + n0 + nl] = acc[v] + b1_s[o];
    }
  }
}

// ---------------------------------------------------------------------------
// Weighted SVD head: one WG per batch; reductions then 3x3 one-sided Jacobi.
// ---------------------------------------------------------------------------
__global__ void __launch_bounds__(256) svd_kernel(
    const float* __restrict__ src, const float* __restrict__ tgt,
    const float* __restrict__ wts, float* __restrict__ out, long obstride) {
  __shared__ float red[256][24];
  const int b = blockIdx.x, tid = threadIdx.x;
  const float* sb = src + (size_t)b * 3 * NP;
  const float* tb = tgt + (size_t)b * 3 * NP;
  const float* wb = wts + (size_t)b * NP;
  float a[22];
#pragma unroll
  for (int i = 0; i < 22; ++i) a[i] = 0.f;
  for (int n = tid; n < NP; n += 256) {
    const float sx = sb[n], sy = sb[NP + n], sz = sb[2 * NP + n];
    const float tx = tb[n], ty = tb[NP + n], tz = tb[2 * NP + n];
    const float w = wb[n];
    a[0] += sx; a[1] += sy; a[2] += sz;
    a[3] += tx; a[4] += ty; a[5] += tz;
    a[6] += w;
    a[7] += w * sx; a[8] += w * sy; a[9] += w * sz;
    a[10] += w * tx; a[11] += w * ty; a[12] += w * tz;
    a[13] += w * sx * tx; a[14] += w * sx * ty; a[15] += w * sx * tz;
    a[16] += w * sy * tx; a[17] += w * sy * ty; a[18] += w * sy * tz;
    a[19] += w * sz * tx; a[20] += w * sz * ty; a[21] += w * sz * tz;
  }
#pragma unroll
  for (int i = 0; i < 22; ++i) red[tid][i] = a[i];
  __syncthreads();
  for (int s = 128; s > 0; s >>= 1) {
    if (tid < s)
      for (int i = 0; i < 22; ++i) red[tid][i] += red[tid + s][i];
    __syncthreads();
  }
  if (tid == 0) {
    float ms[3], mt[3], Sws[3], Swt[3], H[3][3];
    const float Sw = red[0][6];
    for (int i = 0; i < 3; ++i) {
      ms[i]  = red[0][i] * (1.f / NP);
      mt[i]  = red[0][3 + i] * (1.f / NP);
      Sws[i] = red[0][7 + i];
      Swt[i] = red[0][10 + i];
    }
    for (int i = 0; i < 3; ++i)
      for (int j = 0; j < 3; ++j)
        H[i][j] = red[0][13 + i * 3 + j] - ms[i] * Swt[j] - mt[j] * Sws[i] +
                  ms[i] * mt[j] * Sw;
    float A[3][3], V[3][3];
    for (int i = 0; i < 3; ++i)
      for (int j = 0; j < 3; ++j) {
        A[i][j] = H[i][j];
        V[i][j] = (i == j) ? 1.f : 0.f;
      }
    for (int sweep = 0; sweep < 30; ++sweep)
      for (int p = 0; p < 2; ++p)
        for (int q = p + 1; q < 3; ++q) {
          float al = 0.f, be = 0.f, ga = 0.f;
          for (int i = 0; i < 3; ++i) {
            al += A[i][p] * A[i][p];
            be += A[i][q] * A[i][q];
            ga += A[i][p] * A[i][q];
          }
          if (fabsf(ga) <= 1e-12f * sqrtf(al * be)) continue;
          const float ze = (be - al) / (2.f * ga);
          const float t = copysignf(1.f, ze) / (fabsf(ze) + sqrtf(1.f + ze * ze));
          const float c = rsqrtf(1.f + t * t);
          const float s = c * t;
          for (int i = 0; i < 3; ++i) {
            const float ap = A[i][p], aq = A[i][q];
            A[i][p] = c * ap - s * aq; A[i][q] = s * ap + c * aq;
            const float vp = V[i][p], vq = V[i][q];
            V[i][p] = c * vp - s * vq; V[i][q] = s * vp + c * vq;
          }
        }
    float sig[3], U[3][3];
    for (int j = 0; j < 3; ++j) {
      sig[j] = sqrtf(A[0][j] * A[0][j] + A[1][j] * A[1][j] + A[2][j] * A[2][j]);
      const float inv = 1.f / fmaxf(sig[j], 1e-20f);
      for (int i = 0; i < 3; ++i) U[i][j] = A[i][j] * inv;
    }
    const float detU = U[0][0] * (U[1][1] * U[2][2] - U[1][2] * U[2][1]) -
                       U[0][1] * (U[1][0] * U[2][2] - U[1][2] * U[2][0]) +
                       U[0][2] * (U[1][0] * U[2][1] - U[1][1] * U[2][0]);
    const float detV = V[0][0] * (V[1][1] * V[2][2] - V[1][2] * V[2][1]) -
                       V[0][1] * (V[1][0] * V[2][2] - V[1][2] * V[2][0]) +
                       V[0][2] * (V[1][0] * V[2][1] - V[1][1] * V[2][0]);
    if (detU * detV < 0.f) {   // flip V column of smallest sigma (ref's V[:,2])
      int jm = 0;
      if (sig[1] < sig[jm]) jm = 1;
      if (sig[2] < sig[jm]) jm = 2;
      for (int i = 0; i < 3; ++i) V[i][jm] = -V[i][jm];
    }
    float R[3][3];
    for (int i = 0; i < 3; ++i)
      for (int j = 0; j < 3; ++j) {
        float acc = 0.f;
        for (int k = 0; k < 3; ++k) acc += V[i][k] * U[j][k];
        R[i][j] = acc;
      }
    float* ob = out + (size_t)b * obstride + (size_t)EC * NP;
    for (int i = 0; i < 3; ++i) {
      for (int j = 0; j < 3; ++j) ob[i * 3 + j] = R[i][j];
      ob[9 + i] = -(R[i][0] * Sws[0] + R[i][1] * Sws[1] + R[i][2] * Sws[2]) + Swt[i];
    }
  }
}

// ---------------------------------------------------------------------------
// Driver
// ---------------------------------------------------------------------------
extern "C" void kernel_launch(void* const* d_in, const int* in_sizes, int n_in,
                              void* d_out, int out_size, void* d_ws,
                              size_t ws_size, hipStream_t stream) {
  (void)in_sizes; (void)n_in; (void)out_size; (void)ws_size;
  const float* src  = (const float*)d_in[0];
  const float* tgt  = (const float*)d_in[1];
  const float* wts  = (const float*)d_in[2];
  const float* w2a1 = (const float*)d_in[3];
  const float* g1   = (const float*)d_in[4];
  const float* bt1  = (const float*)d_in[5];
  const float* w2b1 = (const float*)d_in[6];
  const float* bb1  = (const float*)d_in[7];
  const float* w1d1 = (const float*)d_in[8];
  const float* b1d1 = (const float*)d_in[9];
  const float* w2a  = (const float*)d_in[10];
  const float* g    = (const float*)d_in[11];
  const float* bt   = (const float*)d_in[12];
  const float* w2b  = (const float*)d_in[13];
  const float* bb   = (const float*)d_in[14];
  const float* w1d  = (const float*)d_in[15];
  const float* b1d  = (const float*)d_in[16];
  float* out = (float*)d_out;

  char* ws = (char*)d_ws;
  size_t off = 0;
  int* nn = (int*)(ws + off);           off += (size_t)NB * NP * KNN * sizeof(int);
  float* xb0 = (float*)(ws + off);      off += (size_t)NB * EC * NP * sizeof(float);
  float* xb1 = (float*)(ws + off);      off += (size_t)NB * EC * NP * sizeof(float);
  float* M1  = (float*)(ws + off);      off += EC * sizeof(float);
  float* M2  = (float*)(ws + off);      off += (size_t)EC * EC * sizeof(float);
  float* scl = (float*)(ws + off);      off += EC * sizeof(float);
  float* bia = (float*)(ws + off);

  const float invS = 1.f / (float)((long)NB * NP * KNN);
  const int nStatBlk = (NB * NP * KNN) / 384;
  const long xstride = (long)EC * NP;
  const long ostride = (long)(EC * NP + 12);

  knn_kernel<<<dim3(NP / 256, NB), 256, 0, stream>>>(src, nn);

  // stage 1 (C = 3, padded to 32)
  zero_kernel<<<(EC + EC * EC + 255) / 256, 256, 0, stream>>>(M1, EC + EC * EC);
  stats_kernel<32><<<dim3(nStatBlk), 128, 0, stream>>>(src, nn, 3, M1, M2);
  finalize_stats_kernel<<<1, EC, 0, stream>>>(M1, M2, w2a1, g1, bt1, 3, invS, scl, bia);
  propagate_kernel<32><<<dim3(NP / 16, NB), 256, PCfg<32>::total, stream>>>(
      src, nn, w2a1, scl, bia, w2b1, bb1, w1d1, b1d1, xb0, xstride, 3);

  // stages 2..5 (C = 64)
  const float* xin = xb0;
  for (int i = 0; i < 4; ++i) {
    const float* w2ai = w2a + (size_t)i * EC * EC;
    const float* gi   = g   + (size_t)i * EC;
    const float* bti  = bt  + (size_t)i * EC;
    const float* w2bi = w2b + (size_t)i * EC * EC;
    const float* bbi  = bb  + (size_t)i * EC;
    const float* w1i  = w1d + (size_t)i * EC * EC;
    const float* b1i  = b1d + (size_t)i * EC;
    float* xo = (i == 3) ? out : ((i & 1) ? xb0 : xb1);
    const long bs = (i == 3) ? ostride : xstride;

    zero_kernel<<<(EC + EC * EC + 255) / 256, 256, 0, stream>>>(M1, EC + EC * EC);
    stats_kernel<64><<<dim3(nStatBlk), 128, 0, stream>>>(xin, nn, EC, M1, M2);
    finalize_stats_kernel<<<1, EC, 0, stream>>>(M1, M2, w2ai, gi, bti, EC, invS, scl, bia);
    propagate_kernel<64><<<dim3(NP / 16, NB), 256, PCfg<64>::total, stream>>>(
        xin, nn, w2ai, scl, bia, w2bi, bbi, w1i, b1i, xo, bs, EC);
    xin = xo;
  }

  svd_kernel<<<NB, 256, 0, stream>>>(src, tgt, wts, out, ostride);
}
